// longsfTSR_35768487641433
// MI455X (gfx1250) — compile-verified, matrix operand order checked by reading, not run on hardware
//
#include <hip/hip_runtime.h>
#include <stdint.h>

// ---------------------------------------------------------------------------
// Problem constants (from the reference)
// ---------------------------------------------------------------------------
constexpr int HH = 256, CL = 256, WL = 512;
constexpr int NTOK = HH * WL;          // 131072 token rows
constexpr int M_MAP = 27648;           // rows per map
constexpr int SEQ = 216;
constexpr int TROWS = 2 * M_MAP;       // 55296 rows per mamba block
constexpr int BATCHES = TROWS / SEQ;   // 256 sequences
constexpr int DI = 512;                // d_inner
constexpr int DT_RANK = 16, D_STATE = 16;
constexpr int DBC_W = DT_RANK + 2 * D_STATE;  // 48

// ---------------------------------------------------------------------------
// WMMA / TDM vector types
// ---------------------------------------------------------------------------
typedef __bf16 v16bf __attribute__((ext_vector_type(16)));
typedef float  v8f   __attribute__((ext_vector_type(8)));
typedef unsigned int v4u __attribute__((ext_vector_type(4)));
typedef int    v4i   __attribute__((ext_vector_type(4)));
typedef int    v8i   __attribute__((ext_vector_type(8)));

union FragU { uint4 q[2]; v16bf v; };

// Load a 16x32 bf16 fragment (A layout; B^T stored (N,K) row-major uses the
// identical lane pattern). Lane l<16: rows M=l, K = k0+[0,8) and k0+[16,24);
// lane l>=16: same rows, K = k0+[8,16) and k0+[24,32). Two b128 loads.
__device__ __forceinline__ v16bf load_frag(const __bf16* __restrict__ base,
                                           int ld, int row0, int k0, int lane) {
  const int r  = row0 + (lane & 15);
  const int kh = k0 + ((lane >> 4) << 3);
  const __bf16* p = base + (size_t)r * ld + kh;
  FragU u;
  u.q[0] = *reinterpret_cast<const uint4*>(p);
  u.q[1] = *reinterpret_cast<const uint4*>(p + 16);
  return u.v;
}

__device__ __forceinline__ float silu_f(float x) {
  return x * (1.0f / (1.0f + __expf(-x)));
}

// ---------------------------------------------------------------------------
// WMMA GEMM:  C[M,N] (f32) = A[M,K] (bf16) * W[N,K]^T (bf16)
// Each wave owns a 32 x (16*NT) strip: 2 A fragments reused across NT columns,
// each B fragment reused across both M tiles -> 2*NT wmma per K step.
// BIAS: add bias[N]; ACT: SiLU epilogue.
// ---------------------------------------------------------------------------
template<int NT, int BIAS, int ACT>
__global__ __launch_bounds__(128) void k_gemm(const __bf16* __restrict__ A,
                                              const __bf16* __restrict__ W,
                                              const float*  __restrict__ bias,
                                              float* __restrict__ C,
                                              int Mdim, int Ndim, int Kdim) {
  const int lane = threadIdx.x & 31;
  const int wave = threadIdx.x >> 5;
  const int m0   = (blockIdx.x * 4 + wave) * 32;
  const int n0   = blockIdx.y * (16 * NT);
  if (m0 >= Mdim) return;

  v8f acc[2][NT] = {};
  for (int k = 0; k < Kdim; k += 32) {
    v16bf a0 = load_frag(A, Kdim, m0,      k, lane);
    v16bf a1 = load_frag(A, Kdim, m0 + 16, k, lane);
    if (k + 64 < Kdim) {  // warm caches ahead (global_prefetch_b8)
      __builtin_prefetch(A + (size_t)(m0 + (lane & 15)) * Kdim + k + 64, 0, 1);
      __builtin_prefetch(A + (size_t)(m0 + 16 + (lane & 15)) * Kdim + k + 64, 0, 1);
    }
#pragma unroll
    for (int t = 0; t < NT; ++t) {
      v16bf b = load_frag(W, Kdim, n0 + 16 * t, k, lane);
      acc[0][t] = __builtin_amdgcn_wmma_f32_16x16x32_bf16(
          false, a0, false, b, (short)0, acc[0][t], false, false);
      acc[1][t] = __builtin_amdgcn_wmma_f32_16x16x32_bf16(
          false, a1, false, b, (short)0, acc[1][t], false, false);
    }
  }
  const int nl = lane & 15;
#pragma unroll
  for (int mh = 0; mh < 2; ++mh) {
    const int mbase = m0 + mh * 16 + ((lane >> 4) << 3);
#pragma unroll
    for (int t = 0; t < NT; ++t) {
      const int n = n0 + 16 * t + nl;
      const float bv = BIAS ? bias[n] : 0.0f;
#pragma unroll
      for (int v = 0; v < 8; ++v) {
        float val = acc[mh][t][v] + bv;
        if (ACT) val = silu_f(val);
        C[(size_t)(mbase + v) * Ndim + n] = val;
      }
    }
  }
}

// ---------------------------------------------------------------------------
// Stage 0: (H,C,W) f32 -> token-major (H*W, C) bf16 via LDS tile transpose
// ---------------------------------------------------------------------------
__global__ __launch_bounds__(256) void k_transpose_bf16(const float* __restrict__ x,
                                                        __bf16* __restrict__ x0) {
  __shared__ float tile[32][33];
  const int h  = blockIdx.z;
  const int w0 = blockIdx.x * 32;
  const int c0 = blockIdx.y * 32;
  const int tx = threadIdx.x, ty = threadIdx.y;  // (32,8)
#pragma unroll
  for (int i = 0; i < 4; ++i) {
    const int c = c0 + ty + i * 8;
    tile[ty + i * 8][tx] = x[((size_t)h * CL + c) * WL + w0 + tx];
  }
  __syncthreads();
#pragma unroll
  for (int i = 0; i < 4; ++i) {
    const int w = w0 + ty + i * 8;
    x0[((size_t)h * WL + w) * CL + c0 + tx] = (__bf16)tile[tx][ty + i * 8];
  }
}

// f32 -> bf16 weight conversion
__global__ void k_cvt_bf16(const float* __restrict__ in, __bf16* __restrict__ out, int n) {
  int i = blockIdx.x * blockDim.x + threadIdx.x;
  if (i < n) out[i] = (__bf16)in[i];
}

// ---------------------------------------------------------------------------
// Gather + RMSNorm: one wave per row (256 elems = 8/lane)
// ---------------------------------------------------------------------------
__global__ __launch_bounds__(256) void k_gather_rms(const float* __restrict__ xf,
                                                    const int* __restrict__ mr,
                                                    const int* __restrict__ mt,
                                                    const float* __restrict__ nw,
                                                    float* __restrict__ raw,
                                                    __bf16* __restrict__ nrm) {
  const int j    = blockIdx.x * 8 + (threadIdx.x >> 5);
  const int lane = threadIdx.x & 31;
  const int src  = (j < M_MAP) ? mr[j] : mt[j - M_MAP];
  const float* rp = xf + (size_t)src * CL + lane * 8;
  float4 v0 = *(const float4*)rp;
  float4 v1 = *(const float4*)(rp + 4);
  float ss = v0.x*v0.x + v0.y*v0.y + v0.z*v0.z + v0.w*v0.w
           + v1.x*v1.x + v1.y*v1.y + v1.z*v1.z + v1.w*v1.w;
#pragma unroll
  for (int m = 16; m >= 1; m >>= 1) ss += __shfl_xor(ss, m, 32);
  const float rs = rsqrtf(ss * (1.0f / 256.0f) + 1e-5f);
  float* wp = raw + (size_t)j * CL + lane * 8;
  *(float4*)wp = v0; *(float4*)(wp + 4) = v1;
  __bf16* np = nrm + (size_t)j * CL + lane * 8;
  const float* g = nw + lane * 8;
  const float vv[8] = {v0.x, v0.y, v0.z, v0.w, v1.x, v1.y, v1.z, v1.w};
#pragma unroll
  for (int i = 0; i < 8; ++i) np[i] = (__bf16)(vv[i] * rs * g[i]);
}

// ---------------------------------------------------------------------------
// Depthwise causal conv(4) + bias + SiLU over xs = xz[:, :512]
// ---------------------------------------------------------------------------
__global__ __launch_bounds__(256) void k_conv_silu(const float* __restrict__ xz,
                                                   const float* __restrict__ cw,
                                                   const float* __restrict__ cb,
                                                   float* __restrict__ xs_f32,
                                                   __bf16* __restrict__ xs_bf) {
  const int idx = blockIdx.x * blockDim.x + threadIdx.x;  // t*512 + ch
  const int t  = idx >> 9;
  const int ch = idx & 511;
  const int p  = t % SEQ;
  float acc = cb[ch];
#pragma unroll
  for (int k = 0; k < 4; ++k) {
    const int q = p - 3 + k;
    if (q >= 0) acc += cw[ch * 4 + k] * xz[(size_t)(t - 3 + k) * 1024 + ch];
  }
  const float v = silu_f(acc);
  xs_f32[idx] = v;
  xs_bf[idx]  = (__bf16)v;
}

// ---------------------------------------------------------------------------
// Selective scan: 1 workgroup per sequence (512 threads = 1 channel each).
// The 216x48 f32 dbc slab (41472 B, contiguous per sequence) is DMA'd into
// LDS by the Tensor Data Mover: wave 0 issues tensor_load_to_lds with a 1-D
// tile descriptor, waits on TENSORcnt, then the workgroup barrier releases
// all 16 waves. Fuses dt=softplus(dbc@dtw+b), the h-recurrence, y+xs*D and
// the SiLU(z) gate; emits bf16 for the out_proj WMMA GEMM.
// ---------------------------------------------------------------------------
__global__ __launch_bounds__(512) void k_scan(const float* __restrict__ dbc,
                                              const float* __restrict__ xs_f32,
                                              const float* __restrict__ xz,
                                              const float* __restrict__ A_log,
                                              const float* __restrict__ dt_w,
                                              const float* __restrict__ dt_b,
                                              const float* __restrict__ Dp,
                                              __bf16* __restrict__ ybf) {
  __shared__ float sd[SEQ * DBC_W];
  const int b  = blockIdx.x;
  const int ch = threadIdx.x;

  if (threadIdx.x < 32) {  // wave 0 drives the TDM
    constexpr unsigned NELEM = (unsigned)(SEQ * DBC_W);  // 10368 f32 elements
    const unsigned lds_addr = (unsigned)(uintptr_t)(&sd[0]);
    const unsigned long long ga =
        (unsigned long long)(uintptr_t)(dbc + (size_t)b * NELEM);
    // ---- D# group 0: count=1 | lds_addr | global_addr[56:0] | type=2
    v4u g0;
    g0[0] = 1u;
    g0[1] = lds_addr;
    g0[2] = (unsigned)(ga & 0xffffffffu);
    g0[3] = (unsigned)((ga >> 32) & 0x01ffffffu) | (2u << 30);
    // ---- D# group 1: data_size=4B; tensor_dim0=tile_dim0=10368 (1-D tile);
    //      tensor_dim1=1; stride0=10368; everything else 0
    v8i g1;
    g1[0] = (int)(2u << 16);                               // data_size = 4B
    g1[1] = (int)((NELEM & 0xffffu) << 16);                // tensor_dim0 lo
    g1[2] = (int)((NELEM >> 16) & 0xffffu) | (1 << 16);    // dim0 hi | tensor_dim1=1
    g1[3] = (int)(NELEM << 16);                            // tile_dim0 = 10368
    g1[4] = 0;                                             // tile_dim1/2 unused
    g1[5] = (int)NELEM;                                    // tensor_dim0_stride
    g1[6] = 0;
    g1[7] = 0;
    v4i gz4 = {0, 0, 0, 0};                                // dims 2..4 unused
    v8i gz8 = {0, 0, 0, 0, 0, 0, 0, 0};
    __builtin_amdgcn_tensor_load_to_lds(g0, g1, gz4, gz4, gz8, 0);
    __builtin_amdgcn_s_wait_tensorcnt(0);
  }
  __syncthreads();

  float Arow[D_STATE], dtw[DT_RANK], h[D_STATE] = {};
#pragma unroll
  for (int s = 0; s < D_STATE; ++s) Arow[s] = -__expf(A_log[ch * D_STATE + s]);
#pragma unroll
  for (int r = 0; r < DT_RANK; ++r) dtw[r] = dt_w[ch * DT_RANK + r];
  const float dtb = dt_b[ch], Dch = Dp[ch];

  for (int t = 0; t < SEQ; ++t) {
    const float* dr = &sd[t * DBC_W];
    float dtv = dtb;
#pragma unroll
    for (int r = 0; r < DT_RANK; ++r) dtv += dr[r] * dtw[r];
    dtv = (dtv > 20.0f) ? dtv : log1pf(__expf(dtv));      // softplus
    const size_t row = (size_t)(b * SEQ + t);
    const float xv  = xs_f32[row * DI + ch];
    const float dtx = dtv * xv;
    float y = 0.0f;
#pragma unroll
    for (int s = 0; s < D_STATE; ++s) {
      const float dA = __expf(dtv * Arow[s]);
      h[s] = dA * h[s] + dtx * dr[DT_RANK + s];
      y += h[s] * dr[DT_RANK + D_STATE + s];
    }
    const float z = xz[row * 1024 + DI + ch];
    ybf[row * DI + ch] = (__bf16)((y + xv * Dch) * silu_f(z));
  }
}

// out_proj result + residual, scattered back into xf
__global__ __launch_bounds__(256) void k_scatter(float* __restrict__ xf,
                                                 const int* __restrict__ mr,
                                                 const int* __restrict__ mt,
                                                 const float* __restrict__ outp,
                                                 const float* __restrict__ raw) {
  const int idx = blockIdx.x * blockDim.x + threadIdx.x;  // j*256 + c
  const int j = idx >> 8, c = idx & 255;
  const int src = (j < M_MAP) ? mr[j] : mt[j - M_MAP];
  xf[(size_t)src * CL + c] = outp[idx] + raw[idx];
}

// LayerNorm + mlp2 (256->1) + sigmoid, one wave per row
__global__ __launch_bounds__(256) void k_score(const float* __restrict__ xf,
                                               const float* __restrict__ g,
                                               const float* __restrict__ bb,
                                               const float* __restrict__ w2,
                                               const float* __restrict__ b2,
                                               float* __restrict__ S) {
  const int j    = blockIdx.x * 8 + (threadIdx.x >> 5);
  const int lane = threadIdx.x & 31;
  const float* rp = xf + (size_t)j * CL + lane * 8;
  float4 v0 = *(const float4*)rp;
  float4 v1 = *(const float4*)(rp + 4);
  const float vv[8] = {v0.x, v0.y, v0.z, v0.w, v1.x, v1.y, v1.z, v1.w};
  float sm = 0.f, ss = 0.f;
#pragma unroll
  for (int i = 0; i < 8; ++i) { sm += vv[i]; ss += vv[i] * vv[i]; }
#pragma unroll
  for (int m = 16; m >= 1; m >>= 1) { sm += __shfl_xor(sm, m, 32); ss += __shfl_xor(ss, m, 32); }
  const float mu = sm * (1.0f / 256.0f);
  const float rs = rsqrtf(ss * (1.0f / 256.0f) - mu * mu + 1e-5f);
  float dot = 0.f;
#pragma unroll
  for (int i = 0; i < 8; ++i) {
    const int c = lane * 8 + i;
    dot += ((vv[i] - mu) * rs * g[c] + bb[c]) * w2[c];
  }
#pragma unroll
  for (int m = 16; m >= 1; m >>= 1) dot += __shfl_xor(dot, m, 32);
  if (lane == 0) S[j] = 1.0f / (1.0f + __expf(-(dot + b2[0])));
}

// out = sigmoid(s)*x + x  (s broadcast over C)
__global__ __launch_bounds__(256) void k_apply(const float* __restrict__ x,
                                               const float* __restrict__ S,
                                               float* __restrict__ out) {
  const int idx = blockIdx.x * blockDim.x + threadIdx.x;  // (h*256+c)*512+w
  const int h = idx >> 17, w = idx & 511;
  out[idx] = x[idx] * (1.0f + S[h * WL + w]);
}

// ---------------------------------------------------------------------------
// Host-side orchestration
// ---------------------------------------------------------------------------
extern "C" void kernel_launch(void* const* d_in, const int* in_sizes, int n_in,
                              void* d_out, int out_size, void* d_ws, size_t ws_size,
                              hipStream_t stream) {
  (void)in_sizes; (void)n_in; (void)out_size; (void)ws_size;
  const float* x       = (const float*)d_in[0];
  const float* mlp1_w  = (const float*)d_in[1];
  const float* mlp1_b  = (const float*)d_in[2];
  const float* ln_g    = (const float*)d_in[23];
  const float* ln_b    = (const float*)d_in[24];
  const float* mlp2_w  = (const float*)d_in[25];
  const float* mlp2_b  = (const float*)d_in[26];

  // Workspace layout
  char* ws = (char*)d_ws;
  size_t off = 0;
  auto alloc = [&](size_t bytes) { size_t r = off; off += (bytes + 255) & ~(size_t)255; return r; };
  float*  XF      = (float*)(ws + alloc((size_t)NTOK * CL * 4));
  __bf16* X0B     = (__bf16*)(ws + alloc((size_t)NTOK * CL * 2));
  float*  FRAW    = (float*)(ws + alloc((size_t)TROWS * CL * 4));
  __bf16* FNRM    = (__bf16*)(ws + alloc((size_t)TROWS * CL * 2));
  float*  XZ      = (float*)(ws + alloc((size_t)TROWS * 1024 * 4));
  float*  XSF     = (float*)(ws + alloc((size_t)TROWS * DI * 4));
  __bf16* XSB     = (__bf16*)(ws + alloc((size_t)TROWS * DI * 2));
  float*  DBC     = (float*)(ws + alloc((size_t)TROWS * DBC_W * 4));
  __bf16* YB      = (__bf16*)(ws + alloc((size_t)TROWS * DI * 2));
  float*  OUTP    = (float*)(ws + alloc((size_t)TROWS * CL * 4));
  float*  SBUF    = (float*)(ws + alloc((size_t)NTOK * 4));
  __bf16* W_MLP1  = (__bf16*)(ws + alloc((size_t)CL * CL * 2));
  __bf16* W_IN[2], *W_XP[2], *W_OP[2];
  for (int i = 0; i < 2; ++i) {
    W_IN[i] = (__bf16*)(ws + alloc((size_t)2 * DI * CL * 2));
    W_XP[i] = (__bf16*)(ws + alloc((size_t)DBC_W * DI * 2));
    W_OP[i] = (__bf16*)(ws + alloc((size_t)CL * DI * 2));
  }

  // Stage 0: transpose + bf16 convert of token matrix
  k_transpose_bf16<<<dim3(WL / 32, CL / 32, HH), dim3(32, 8), 0, stream>>>(x, X0B);

  // Weight conversions (tiny, L2-resident afterwards)
  auto cvt = [&](const float* src, __bf16* dst, int n) {
    k_cvt_bf16<<<(n + 255) / 256, 256, 0, stream>>>(src, dst, n);
  };
  cvt(mlp1_w, W_MLP1, CL * CL);
  for (int i = 0; i < 2; ++i) {
    cvt((const float*)d_in[4 + 10 * i],  W_IN[i], 2 * DI * CL);
    cvt((const float*)d_in[7 + 10 * i],  W_XP[i], DBC_W * DI);
    cvt((const float*)d_in[12 + 10 * i], W_OP[i], CL * DI);
  }

  // Stage 1: mlp1 (131072x256x256) + bias + SiLU -> XF (f32)
  k_gemm<4, 1, 1><<<dim3(NTOK / 128, CL / 64), 128, 0, stream>>>(
      X0B, W_MLP1, mlp1_b, XF, NTOK, CL, CL);

  // Stage 2: two Mamba blocks
  for (int i = 0; i < 2; ++i) {
    const float* norm_w = (const float*)d_in[3 + 10 * i];
    const float* conv_w = (const float*)d_in[5 + 10 * i];
    const float* conv_b = (const float*)d_in[6 + 10 * i];
    const float* dt_w   = (const float*)d_in[8 + 10 * i];
    const float* dt_b   = (const float*)d_in[9 + 10 * i];
    const float* A_log  = (const float*)d_in[10 + 10 * i];
    const float* Dp     = (const float*)d_in[11 + 10 * i];
    const int* mr = (const int*)d_in[27 + 2 * i];
    const int* mt = (const int*)d_in[28 + 2 * i];

    k_gather_rms<<<TROWS / 8, 256, 0, stream>>>(XF, mr, mt, norm_w, FRAW, FNRM);
    // in_proj: 55296 x 256 -> 1024
    k_gemm<4, 0, 0><<<dim3(TROWS / 128, 1024 / 64), 128, 0, stream>>>(
        FNRM, W_IN[i], nullptr, XZ, TROWS, 1024, CL);
    k_conv_silu<<<(TROWS * DI) / 256, 256, 0, stream>>>(XZ, conv_w, conv_b, XSF, XSB);
    // x_proj: 55296 x 512 -> 48
    k_gemm<3, 0, 0><<<dim3(TROWS / 128, 1), 128, 0, stream>>>(
        XSB, W_XP[i], nullptr, DBC, TROWS, DBC_W, DI);
    k_scan<<<BATCHES, DI, 0, stream>>>(DBC, XSF, XZ, A_log, dt_w, dt_b, Dp, YB);
    // out_proj: 55296 x 512 -> 256
    k_gemm<4, 0, 0><<<dim3(TROWS / 128, CL / 64), 128, 0, stream>>>(
        YB, W_OP[i], nullptr, OUTP, TROWS, CL, DI);
    k_scatter<<<(TROWS * CL) / 256, 256, 0, stream>>>(XF, mr, mt, OUTP, FRAW);
  }

  // Stage 3: LN + mlp2 + sigmoid gate
  k_score<<<NTOK / 8, 256, 0, stream>>>(XF, ln_g, ln_b, mlp2_w, mlp2_b, SBUF);
  k_apply<<<(HH * CL * WL) / 256, 256, 0, stream>>>(x, SBUF, (float*)d_out);
}
